// VQVAE_26027501814562
// MI455X (gfx1250) — matrix operand order repaired and checked by reading, not compiled, not run on hardware
//
#include <hip/hip_runtime.h>
#include <math.h>

// ---------------------------------------------------------------------------
// Types for CDNA5 WMMA (wave32): v_wmma_f32_16x16x32_bf16
// ---------------------------------------------------------------------------
typedef __attribute__((ext_vector_type(16))) __bf16 v16bf;
typedef __attribute__((ext_vector_type(8)))  float  v8f;

union BFV {
    v16bf          v;
    uint4          q[2];
    unsigned short h[16];
};

__device__ __forceinline__ unsigned short f2bf(float f) {
    union { float f; unsigned u; } x; x.f = f;
    unsigned r = x.u + 0x7FFFu + ((x.u >> 16) & 1u);   // RNE
    return (unsigned short)(r >> 16);
}
__device__ __forceinline__ float bf2f(unsigned short b) {
    union { unsigned u; float f; } x; x.u = ((unsigned)b) << 16;
    return x.f;
}

// ---------------------------------------------------------------------------
// Implicit-GEMM convolution (and ConvTranspose via lhs-dilation) with WMMA.
// Activations: NHWC bf16 (channel stride inCt, slice offset inoff).
// Weights:     repacked [Cout][Kp] bf16, K-order = (kh, kw, ci), Kp = K pad32.
// One wave computes a 16(cout) x 64(pixel) tile = 4 WMMA accumulators: the A
// (weight) fragment is loaded once per K-step and reused by 4 back-to-back
// v_wmma issues (independent D regs -> no RAW hazard stalls), amortizing the
// weight traffic 4x. 8 waves per block (fills a WGP under wave32).
// Epilogue fuses: *scale (BN), +bias, +residual, ReLU/SiLU, bf16-NHWC store
// (with channel offset for concat targets) or f32-NCHW store (final output).
// ---------------------------------------------------------------------------
#define NPW 4   // pixel tiles per wave

__global__ __launch_bounds__(256) void conv_wmma(
    const unsigned short* __restrict__ X,
    const unsigned short* __restrict__ Wr,
    const float*          __restrict__ Bias,   // may be null
    const unsigned short* __restrict__ Res,    // may be null (packed Cout NHWC)
    unsigned short*       __restrict__ Y,      // bf16 NHWC out (if Yf == null)
    float*                __restrict__ Yf,     // f32 NCHW out (final conv)
    int N, int H, int W, int Cin, int inCt, int inoff,
    int OH, int OW, int Cout, int outCt, int outoff,
    int KH, int KW, int stride, int pad, int ldil,
    int K, int Kp, int cinSh, float scale, int act, int fast)
{
    const int lane = threadIdx.x & 31;
    const int wave = threadIdx.x >> 5;
    const int m    = lane & 15;     // A row / B col / D col index
    const int kh2  = lane >> 4;     // K-half selector
    const int tp   = blockIdx.x * 8 + wave;
    const int tc   = blockIdx.y;

    // pixel coordinates for the NPW B-operand tiles
    int pixA[NPW], ohA[NPW], owA[NPW], nbA[NPW];
#pragma unroll
    for (int t = 0; t < NPW; ++t) {
        int pix = tp * (16 * NPW) + t * 16 + m;
        pixA[t] = pix;
        owA[t] = pix % OW; int q = pix / OW;
        ohA[t] = q % OH;   nbA[t] = q / OH;
    }

    const unsigned short* wrow = Wr + (size_t)(tc * 16 + m) * Kp;

    v8f acc[NPW];
#pragma unroll
    for (int t = 0; t < NPW; ++t) acc[t] = (v8f){0.f,0.f,0.f,0.f,0.f,0.f,0.f,0.f};

    for (int k0 = 0; k0 < Kp; k0 += 32) {
        BFV a;
        // A operand (weights): per-lane run [k0+kh2*16,+16)
        a.q[0] = *(const uint4*)(wrow + k0 + kh2 * 16);
        a.q[1] = *(const uint4*)(wrow + k0 + kh2 * 16 + 8);
        __builtin_prefetch(wrow + k0 + 64, 0, 1);   // speculative: next K chunks

        const int ks = k0 + kh2 * 16;   // B per-lane run [ks, ks+16)
        if (fast) {   // Cin is a power of two and multiple of 16
            int seg = ks >> cinSh;
            int ci0 = ks & (Cin - 1);
            int kw = seg % KW, kh = seg / KW;
            const int ksOK = (ks < K);
#pragma unroll
            for (int t = 0; t < NPW; ++t) {
                BFV b;
                int ih, iw; int ok = ksOK;
                if (ldil == 1) { ih = ohA[t] * stride - pad + kh; iw = owA[t] * stride - pad + kw; }
                else { int ihv = ohA[t] - pad + kh, iwv = owA[t] - pad + kw;   // lhs_dilation=2
                       ok &= (((ihv | iwv) & 1) == 0); ih = ihv >> 1; iw = iwv >> 1; }
                ok &= (ih >= 0) & (ih < H) & (iw >= 0) & (iw < W);
                if (ok) {
                    const uint4* bp = (const uint4*)(X + ((size_t)(nbA[t] * H + ih) * W + iw) * inCt + inoff + ci0);
                    b.q[0] = bp[0]; b.q[1] = bp[1];
                } else {
                    b.q[0] = make_uint4(0u, 0u, 0u, 0u);
                    b.q[1] = make_uint4(0u, 0u, 0u, 0u);
                }
                acc[t] = __builtin_amdgcn_wmma_f32_16x16x32_bf16(false, a.v, false, b.v,
                                                                 (short)0, acc[t], false, false);
            }
        } else {      // generic gather (tiny-Cin stem / remixer convs)
#pragma unroll
            for (int t = 0; t < NPW; ++t) {
                BFV b;
                for (int j = 0; j < 16; ++j) {
                    int k = ks + j; unsigned short v = 0;
                    if (k < K) {
                        int ci = k % Cin; int s2 = k / Cin;
                        int kw = s2 % KW, kh = s2 / KW;
                        int ih, iw; int ok = 1;
                        if (ldil == 1) { ih = ohA[t] * stride - pad + kh; iw = owA[t] * stride - pad + kw; }
                        else { int ihv = ohA[t] - pad + kh, iwv = owA[t] - pad + kw;
                               ok = (((ihv | iwv) & 1) == 0); ih = ihv >> 1; iw = iwv >> 1; }
                        if (ok && ih >= 0 && ih < H && iw >= 0 && iw < W)
                            v = X[((size_t)(nbA[t] * H + ih) * W + iw) * inCt + inoff + ci];
                    }
                    b.h[j] = v;
                }
                acc[t] = __builtin_amdgcn_wmma_f32_16x16x32_bf16(false, a.v, false, b.v,
                                                                 (short)0, acc[t], false, false);
            }
        }
    }

    // Epilogue: D element r in this lane -> (co = tc*16 + r + 8*kh2, pixel pixA[t])
    const int cobase = tc * 16 + 8 * kh2;
#pragma unroll
    for (int t = 0; t < NPW; ++t) {
        BFV rb, ob;
        if (Res) rb.q[0] = *(const uint4*)(Res + (size_t)pixA[t] * Cout + cobase);
        for (int r = 0; r < 8; ++r) {
            float v = acc[t][r] * scale;
            if (Bias) v += Bias[cobase + r];
            if (Res)  v += bf2f(rb.h[r]);
            if (act == 1)      v = fmaxf(v, 0.f);
            else if (act == 2) v = v / (1.f + __expf(-v));   // SiLU
            if (Yf) Yf[(((size_t)nbA[t] * Cout + cobase + r) * OH + ohA[t]) * OW + owA[t]] = v;
            else    ob.h[r] = f2bf(v);
        }
        if (!Yf) *(uint4*)(Y + (size_t)pixA[t] * outCt + outoff + cobase) = ob.q[0];
    }
}

// ---------------------------------------------------------------------------
// Weight repack: OIHW f32 -> [Cout][Kp] bf16, K-order (kh,kw,ci), zero-padded.
// isT=1: source is ConvTranspose weight (Cin,Cout,k,k); apply swapaxes+rot180.
// ---------------------------------------------------------------------------
__global__ void repack_w(const float* __restrict__ S, unsigned short* __restrict__ D,
                         int Cout, int Cin, int KH, int KW, int Kp, int isT)
{
    int k = blockIdx.x * 256 + threadIdx.x;
    int o = blockIdx.y;
    if (k >= Kp) return;
    int K = KH * KW * Cin;
    float f = 0.f;
    if (k < K) {
        int ci = k % Cin; int s = k / Cin;
        int kw = s % KW, kh = s / KW;
        if (isT) f = S[(((size_t)ci * Cout + o) * KH + (KH - 1 - kh)) * KW + (KW - 1 - kw)];
        else     f = S[(((size_t)o * Cin + ci) * KH + kh) * KW + kw];
    }
    D[(size_t)o * Kp + k] = f2bf(f);
}

__global__ void nchw2nhwc(const float* __restrict__ X, unsigned short* __restrict__ Y,
                          int N, int C, int H, int W)
{
    size_t i = (size_t)blockIdx.x * 256 + threadIdx.x;
    size_t tot = (size_t)N * C * H * W;
    if (i >= tot) return;
    int c = (int)(i % C); size_t p = i / C;
    int w = (int)(p % W); p /= W;
    int h = (int)(p % H); int nb = (int)(p / H);
    Y[i] = f2bf(X[(((size_t)nb * C + c) * H + h) * W + w]);
}

__global__ void maxpool5_k(const unsigned short* __restrict__ X, unsigned short* __restrict__ Y,
                           int N, int H, int W, int Ct, int ci, int co, int C)
{
    int i = blockIdx.x * 256 + threadIdx.x;
    int tot = N * H * W * C;
    if (i >= tot) return;
    int c = i % C; int p = i / C;
    int w = p % W; p /= W;
    int h = p % H; int nb = p / H;
    float mx = -3.0e38f;
    for (int dh = -2; dh <= 2; ++dh)
        for (int dw = -2; dw <= 2; ++dw) {
            int ih = h + dh, iw = w + dw;
            if (ih >= 0 && ih < H && iw >= 0 && iw < W)
                mx = fmaxf(mx, bf2f(X[((size_t)(nb * H + ih) * W + iw) * Ct + ci + c]));
        }
    Y[((size_t)(nb * H + h) * W + w) * Ct + co + c] = f2bf(mx);
}

// ---------------------------------------------------------------------------
// VQ: codebook prep, fused GEMM(-2 z.E^T)+argmin via WMMA, gather+loss+hist.
// ---------------------------------------------------------------------------
__global__ void cb_prep(const float* __restrict__ E, unsigned short* __restrict__ Eb,
                        float* __restrict__ En2, int K, int D)
{
    int k = blockIdx.x * 64 + threadIdx.x;
    if (k >= K) return;
    float s = 0.f;
    for (int c = 0; c < D; ++c) {
        float v = E[(size_t)k * D + c];
        s += v * v;
        Eb[(size_t)k * D + c] = f2bf(v);
    }
    En2[k] = s;
}

__global__ __launch_bounds__(256) void vq_argmin(
    const unsigned short* __restrict__ Z,   // [P][64] bf16
    const unsigned short* __restrict__ Eb,  // [512][64] bf16
    const float* __restrict__ En2, int* __restrict__ Idx, int P)
{
    const int lane = threadIdx.x & 31, wave = threadIdx.x >> 5;
    const int tp = blockIdx.x * 8 + wave;
    const int m = lane & 15, kh2 = lane >> 4;

    const unsigned short* zrow = Z + (size_t)(tp * 16 + m) * 64;
    BFV a0, a1;
    a0.q[0] = *(const uint4*)(zrow + kh2 * 16);
    a0.q[1] = *(const uint4*)(zrow + kh2 * 16 + 8);
    a1.q[0] = *(const uint4*)(zrow + 32 + kh2 * 16);
    a1.q[1] = *(const uint4*)(zrow + 32 + kh2 * 16 + 8);

    float bval[8]; int bidx[8];
    for (int r = 0; r < 8; ++r) { bval[r] = 3.4e38f; bidx[r] = 0; }

    for (int kt = 0; kt < 32; ++kt) {              // 512 codebook entries / 16
        const unsigned short* erow = Eb + (size_t)(kt * 16 + m) * 64;
        BFV b0, b1;
        b0.q[0] = *(const uint4*)(erow + kh2 * 16);
        b0.q[1] = *(const uint4*)(erow + kh2 * 16 + 8);
        b1.q[0] = *(const uint4*)(erow + 32 + kh2 * 16);
        b1.q[1] = *(const uint4*)(erow + 32 + kh2 * 16 + 8);
        v8f acc = {0.f, 0.f, 0.f, 0.f, 0.f, 0.f, 0.f, 0.f};
        acc = __builtin_amdgcn_wmma_f32_16x16x32_bf16(false, a0.v, false, b0.v, (short)0, acc, false, false);
        acc = __builtin_amdgcn_wmma_f32_16x16x32_bf16(false, a1.v, false, b1.v, (short)0, acc, false, false);
        const int entry = kt * 16 + m;             // N = lane&15
        const float en = En2[entry];
        for (int r = 0; r < 8; ++r) {
            float d = en - 2.f * acc[r];           // |z|^2 constant per row
            if (d < bval[r] || (d == bval[r] && entry < bidx[r])) { bval[r] = d; bidx[r] = entry; }
        }
    }
    // reduce across the 16 lanes that share each pixel (same kh2 half)
    for (int o = 1; o < 16; o <<= 1)
        for (int r = 0; r < 8; ++r) {
            float ov = __shfl_xor(bval[r], o, 32);
            int   oi = __shfl_xor(bidx[r], o, 32);
            if (ov < bval[r] || (ov == bval[r] && oi < bidx[r])) { bval[r] = ov; bidx[r] = oi; }
        }
    if (m == 0)
        for (int r = 0; r < 8; ++r) Idx[tp * 16 + r + 8 * kh2] = bidx[r];
}

__global__ void vq_gather(const unsigned short* __restrict__ Z, const float* __restrict__ E,
                          const int* __restrict__ Idx, unsigned short* __restrict__ ZQ,
                          float* __restrict__ acc, float* __restrict__ hist, int P, int D)
{
    int p = blockIdx.x * 256 + threadIdx.x;
    if (p >= P) return;
    int id = Idx[p];
    float s = 0.f;
    for (int c = 0; c < D; ++c) {
        float q = E[(size_t)id * D + c];
        float z = bf2f(Z[(size_t)p * D + c]);
        float d = q - z; s += d * d;
        ZQ[(size_t)p * D + c] = f2bf(q);           // straight-through fwd value
    }
    atomicAdd(acc, s);
    atomicAdd(&hist[id], 1.0f);
}

__global__ void vq_final(const float* __restrict__ acc, const float* __restrict__ hist,
                         float* __restrict__ out, int outsz, int P, int D, int K)
{
    __shared__ float sh[512];
    int t = threadIdx.x;
    float s = 0.f;
    for (int k = t; k < K; k += blockDim.x) {
        float a = hist[k] / (float)P;
        s += a * logf(a + 1e-10f);
    }
    sh[t] = s; __syncthreads();
    for (int o = blockDim.x >> 1; o > 0; o >>= 1) {
        if (t < o) sh[t] += sh[t + o];
        __syncthreads();
    }
    if (t == 0) {
        out[outsz - 2] = 1.25f * acc[0] / (float)((size_t)P * D);  // q_loss + 0.25*e_loss
        out[outsz - 1] = expf(-sh[0]);                             // perplexity
    }
}

__global__ void zero_f(float* p, int n)
{
    int i = blockIdx.x * 256 + threadIdx.x;
    if (i < n) p[i] = 0.f;
}

// ---------------------------------------------------------------------------
// Host orchestration
// ---------------------------------------------------------------------------
static inline void launch_conv(hipStream_t st,
    const unsigned short* X, const unsigned short* Wr, const float* Bias,
    const unsigned short* Res, unsigned short* Y, float* Yf,
    int N, int H, int W, int Cin, int inCt, int inoff,
    int Cout, int outCt, int outoff,
    int KH, int KW, int stride, int pad, int ldil, float scale, int act)
{
    int OH, OW;
    if (ldil == 2) { OH = 2 * H; OW = 2 * W; }
    else { OH = (H + 2 * pad - KH) / stride + 1; OW = (W + 2 * pad - KW) / stride + 1; }
    int K = KH * KW * Cin, Kp = (K + 31) & ~31;
    int P = N * OH * OW;
    int fast = ((Cin & 15) == 0 && (Cin & (Cin - 1)) == 0) ? 1 : 0;
    int cinSh = fast ? __builtin_ctz((unsigned)Cin) : 0;
    dim3 g((unsigned)(P / (128 * NPW)), (unsigned)(Cout / 16), 1);   // 8 waves x 64 px
    conv_wmma<<<g, dim3(256, 1, 1), 0, st>>>(X, Wr, Bias, Res, Y, Yf,
        N, H, W, Cin, inCt, inoff, OH, OW, Cout, outCt, outoff,
        KH, KW, stride, pad, ldil, K, Kp, cinSh, scale, act, fast);
}

extern "C" void kernel_launch(void* const* d_in, const int* in_sizes, int n_in,
                              void* d_out, int out_size, void* d_ws, size_t ws_size,
                              hipStream_t stream)
{
    (void)in_sizes; (void)n_in; (void)ws_size;
    const int NB = 4;
    const float BN = 0.9999950000374996f;   // 1/sqrt(1+1e-5)

    size_t off = 0;
    auto alloc = [&](size_t bytes) -> void* {
        off = (off + 255) & ~(size_t)255;
        void* p = (char*)d_ws + off; off += bytes; return p;
    };
    auto aU = [&](size_t n) { return (unsigned short*)alloc(n * 2); };
    auto aF = [&](size_t n) { return (float*)alloc(n * 4); };

    // ---- activation buffers (NHWC bf16) ----
    unsigned short* xh   = aU(4ull * 512 * 512 * 3);
    unsigned short* a1   = aU(4ull * 256 * 256 * 32);
    unsigned short* a2   = aU(4ull * 128 * 128 * 64);
    unsigned short* tA   = aU(4ull * 128 * 128 * 64);
    unsigned short* tB   = aU(4ull * 128 * 128 * 64);
    unsigned short* cat7 = aU(4ull * 64 * 64 * 256);
    unsigned short* t3a  = aU(4ull * 64 * 64 * 128);
    unsigned short* t3b  = aU(4ull * 64 * 64 * 128);
    unsigned short* t3c  = aU(4ull * 64 * 64 * 128);
    unsigned short* cat6 = aU(4ull * 32 * 32 * 512);
    unsigned short* t4a  = aU(4ull * 32 * 32 * 256);
    unsigned short* t4b  = aU(4ull * 32 * 32 * 256);
    unsigned short* t4c  = aU(4ull * 32 * 32 * 256);
    unsigned short* t5a  = aU(4ull * 16 * 16 * 256);
    unsigned short* t5b  = aU(4ull * 16 * 16 * 256);
    unsigned short* t5c  = aU(4ull * 16 * 16 * 256);
    unsigned short* sppf = aU(4ull * 16 * 16 * 512);
    unsigned short* z1   = aU(4ull * 64 * 64 * 64);
    unsigned short* z2   = aU(4ull * 64 * 64 * 64);
    unsigned short* z3   = aU(4ull * 64 * 64 * 64);
    unsigned short* dA   = aU(4ull * 64 * 64 * 128);
    unsigned short* dB   = aU(4ull * 64 * 64 * 128);
    unsigned short* dC   = aU(4ull * 64 * 64 * 128);
    unsigned short* r16  = aU(4ull * 512 * 512 * 16);
    unsigned short* c32  = aU(4ull * 512 * 512 * 32);
    unsigned short* r32  = aU(4ull * 512 * 512 * 32);
    unsigned short* Eb   = aU(512 * 64);
    float* En2   = aF(512);
    float* vqacc = aF(1);
    float* hist  = aF(512);
    int*   Idx   = (int*)alloc(16384 * 4);

    // ---- weight repack (OIHW f32 -> [O][Kp] bf16) ----
    auto RW = [&](int idx, int Cout, int Cin, int KH, int KW, int isT) -> unsigned short* {
        int K = KH * KW * Cin, Kp = (K + 31) & ~31;
        unsigned short* dst = aU((size_t)Cout * Kp);
        dim3 g((unsigned)((Kp + 255) / 256), (unsigned)Cout);
        repack_w<<<g, 256, 0, stream>>>((const float*)d_in[idx], dst, Cout, Cin, KH, KW, Kp, isT);
        return dst;
    };
    unsigned short* w_c1a = RW(1, 32, 3, 4, 4, 0);
    unsigned short* w_c1b = RW(2, 64, 32, 4, 4, 0);
    unsigned short* r2w[4]; for (int i = 0; i < 4; ++i) r2w[i] = RW(3 + i, 64, 64, 3, 3, 0);
    unsigned short* r3b0w1 = RW(7, 128, 64, 3, 3, 0), *r3b0w2 = RW(8, 128, 128, 3, 3, 0), *r3b0wd = RW(9, 128, 64, 1, 1, 0);
    unsigned short* r3b1w1 = RW(10, 128, 128, 3, 3, 0), *r3b1w2 = RW(11, 128, 128, 3, 3, 0);
    unsigned short* r3b2w1 = RW(12, 128, 128, 3, 3, 0), *r3b2w2 = RW(13, 128, 128, 3, 3, 0);
    unsigned short* r4b0w1 = RW(14, 256, 128, 3, 3, 0), *r4b0w2 = RW(15, 256, 256, 3, 3, 0), *r4b0wd = RW(16, 256, 128, 1, 1, 0);
    unsigned short* r4b1w1 = RW(17, 256, 256, 3, 3, 0), *r4b1w2 = RW(18, 256, 256, 3, 3, 0);
    unsigned short* r4b2w1 = RW(19, 256, 256, 3, 3, 0), *r4b2w2 = RW(20, 256, 256, 3, 3, 0);
    unsigned short* r4b3w1 = RW(21, 256, 256, 3, 3, 0), *r4b3w2 = RW(22, 256, 256, 3, 3, 0);
    unsigned short* r5b0w1 = RW(23, 256, 256, 3, 3, 0), *r5b0w2 = RW(24, 256, 256, 3, 3, 0), *r5b0wd = RW(25, 256, 256, 1, 1, 0);
    unsigned short* r5b1w1 = RW(26, 256, 256, 3, 3, 0), *r5b1w2 = RW(27, 256, 256, 3, 3, 0);
    unsigned short* w_sp1 = RW(28, 128, 256, 1, 1, 0);
    unsigned short* w_sp2 = RW(29, 256, 512, 1, 1, 0);
    unsigned short* w_c6u = RW(30, 256, 256, 4, 4, 1);
    unsigned short* w_c6c = RW(32, 256, 512, 1, 1, 0);
    unsigned short* r6b0w1 = RW(33, 128, 256, 3, 3, 0), *r6b0w2 = RW(34, 128, 128, 3, 3, 0), *r6b0wd = RW(35, 128, 256, 1, 1, 0);
    unsigned short* r6b1w1 = RW(36, 128, 128, 3, 3, 0), *r6b1w2 = RW(37, 128, 128, 3, 3, 0);
    unsigned short* w_c7u = RW(38, 128, 128, 4, 4, 1);
    unsigned short* w_c7c = RW(40, 128, 256, 1, 1, 0);
    unsigned short* r7b0w1 = RW(41, 64, 128, 3, 3, 0), *r7b0w2 = RW(42, 64, 64, 3, 3, 0), *r7b0wd = RW(43, 64, 128, 1, 1, 0);
    unsigned short* r7b1w1 = RW(44, 64, 64, 3, 3, 0), *r7b1w2 = RW(45, 64, 64, 3, 3, 0);
    unsigned short* w_din = RW(47, 128, 64, 1, 1, 0);
    unsigned short* drw[4]; for (int i = 0; i < 4; ++i) drw[i] = RW(48 + i, 128, 128, 3, 3, 0);
    unsigned short* w_dt1 = RW(52, 64, 128, 4, 4, 1);
    unsigned short* w_dt2 = RW(54, 32, 64, 4, 4, 1);
    unsigned short* w_dt3 = RW(56, 16, 32, 4, 4, 1);
    unsigned short* w_rm1 = RW(58, 16, 3, 1, 1, 0);
    unsigned short* w_rm2 = RW(59, 16, 16, 1, 1, 0);
    unsigned short* w_rm3 = RW(60, 32, 32, 1, 1, 0);
    unsigned short* w_rm4 = RW(61, 16, 32, 1, 1, 0);

    // ---- residual block helper: relu(BN*conv(relu(BN*conv(x))) + idn) ----
    auto RB = [&](const unsigned short* Xin, int H, int W, int Cin, int inCt,
                  int Cout, int s,
                  unsigned short* w1, unsigned short* w2, unsigned short* wd,
                  unsigned short* tIdn, unsigned short* tO1,
                  unsigned short* outB, int outCt, int outoff) {
        int OH = H / s, OW = W / s;
        const unsigned short* resp;
        if (wd) {
            launch_conv(stream, Xin, wd, nullptr, nullptr, tIdn, nullptr,
                        NB, H, W, Cin, inCt, 0, Cout, Cout, 0, 1, 1, s, 0, 1, BN, 0);
            resp = tIdn;
        } else resp = Xin;
        launch_conv(stream, Xin, w1, nullptr, nullptr, tO1, nullptr,
                    NB, H, W, Cin, inCt, 0, Cout, Cout, 0, 3, 3, s, 1, 1, BN, 1);
        launch_conv(stream, tO1, w2, nullptr, resp, outB, nullptr,
                    NB, OH, OW, Cout, Cout, 0, Cout, outCt, outoff, 3, 3, 1, 1, 1, BN, 1);
    };

    // ================= Encoder =================
    {
        size_t tot = 4ull * 512 * 512 * 3;
        nchw2nhwc<<<(unsigned)((tot + 255) / 256), 256, 0, stream>>>((const float*)d_in[0], xh, NB, 3, 512, 512);
    }
    launch_conv(stream, xh, w_c1a, nullptr, nullptr, a1, nullptr, NB, 512, 512, 3, 3, 0, 32, 32, 0, 4, 4, 2, 1, 1, 1.f, 1);
    launch_conv(stream, a1, w_c1b, nullptr, nullptr, a2, nullptr, NB, 256, 256, 32, 32, 0, 64, 64, 0, 4, 4, 2, 1, 1, 1.f, 1);
    // r2
    RB(a2, 128, 128, 64, 64, 64, 1, r2w[0], r2w[1], nullptr, nullptr, tA, tB, 64, 0);
    RB(tB, 128, 128, 64, 64, 64, 1, r2w[2], r2w[3], nullptr, nullptr, tA, a2, 64, 0);     // p2 = a2
    // r3 -> p3 lands in cat7[:,0:128]
    RB(a2, 128, 128, 64, 64, 128, 2, r3b0w1, r3b0w2, r3b0wd, t3a, t3b, t3c, 128, 0);
    RB(t3c, 64, 64, 128, 128, 128, 1, r3b1w1, r3b1w2, nullptr, nullptr, t3a, t3b, 128, 0);
    RB(t3b, 64, 64, 128, 128, 128, 1, r3b2w1, r3b2w2, nullptr, nullptr, t3a, cat7, 256, 0);
    // r4 (reads strided p3) -> p4 lands in cat6[:,0:256]
    RB(cat7, 64, 64, 128, 256, 256, 2, r4b0w1, r4b0w2, r4b0wd, t4a, t4b, t4c, 256, 0);
    RB(t4c, 32, 32, 256, 256, 256, 1, r4b1w1, r4b1w2, nullptr, nullptr, t4a, t4b, 256, 0);
    RB(t4b, 32, 32, 256, 256, 256, 1, r4b2w1, r4b2w2, nullptr, nullptr, t4a, t4c, 256, 0);
    RB(t4c, 32, 32, 256, 256, 256, 1, r4b3w1, r4b3w2, nullptr, nullptr, t4a, cat6, 512, 0);
    // r5 (reads strided p4)
    RB(cat6, 32, 32, 256, 512, 256, 2, r5b0w1, r5b0w2, r5b0wd, t5a, t5b, t5c, 256, 0);
    RB(t5c, 16, 16, 256, 256, 256, 1, r5b1w1, r5b1w2, nullptr, nullptr, t5a, t5b, 256, 0); // p5 = t5b
    // SPPF: cv1 into sppf[:,0:128]; 3 chained pools into slices; cv2 -> t5c
    launch_conv(stream, t5b, w_sp1, nullptr, nullptr, sppf, nullptr, NB, 16, 16, 256, 256, 0, 128, 512, 0, 1, 1, 1, 0, 1, 1.f, 0);
    {
        int tot = NB * 16 * 16 * 128;
        unsigned g = (unsigned)((tot + 255) / 256);
        maxpool5_k<<<g, 256, 0, stream>>>(sppf, sppf, NB, 16, 16, 512, 0,   128, 128);
        maxpool5_k<<<g, 256, 0, stream>>>(sppf, sppf, NB, 16, 16, 512, 128, 256, 128);
        maxpool5_k<<<g, 256, 0, stream>>>(sppf, sppf, NB, 16, 16, 512, 256, 384, 128);
    }
    launch_conv(stream, sppf, w_sp2, nullptr, nullptr, t5c, nullptr, NB, 16, 16, 512, 512, 0, 256, 256, 0, 1, 1, 1, 0, 1, 1.f, 0);
    // FPN cat6: convT(p5) into cat6[:,256:512]; 1x1 -> p6c
    launch_conv(stream, t5c, w_c6u, (const float*)d_in[31], nullptr, cat6, nullptr,
                NB, 16, 16, 256, 256, 0, 256, 512, 256, 4, 4, 2, 2, 2, 1.f, 0);
    launch_conv(stream, cat6, w_c6c, nullptr, nullptr, t4a, nullptr, NB, 32, 32, 512, 512, 0, 256, 256, 0, 1, 1, 1, 0, 1, 1.f, 0);
    // r6
    RB(t4a, 32, 32, 256, 256, 128, 1, r6b0w1, r6b0w2, r6b0wd, t4b, t4c, t3a, 128, 0);
    RB(t3a, 32, 32, 128, 128, 128, 1, r6b1w1, r6b1w2, nullptr, nullptr, t4c, t3b, 128, 0); // p6 = t3b
    // FPN cat7: convT(p6) into cat7[:,128:256]; 1x1 -> p7c
    launch_conv(stream, t3b, w_c7u, (const float*)d_in[39], nullptr, cat7, nullptr,
                NB, 32, 32, 128, 128, 0, 128, 256, 128, 4, 4, 2, 2, 2, 1.f, 0);
    launch_conv(stream, cat7, w_c7c, nullptr, nullptr, t3c, nullptr, NB, 64, 64, 256, 256, 0, 128, 128, 0, 1, 1, 1, 0, 1, 1.f, 0);
    // r7
    RB(t3c, 64, 64, 128, 128, 64, 1, r7b0w1, r7b0w2, r7b0wd, z1, z2, z3, 64, 0);
    RB(z3, 64, 64, 64, 64, 64, 1, r7b1w1, r7b1w2, nullptr, nullptr, z1, z2, 64, 0);        // p7 = z2

    // ================= Vector quantizer =================
    cb_prep<<<8, 64, 0, stream>>>((const float*)d_in[46], Eb, En2, 512, 64);
    vq_argmin<<<128, 256, 0, stream>>>(z2, Eb, En2, Idx, 16384);
    zero_f<<<1, 256, 0, stream>>>(vqacc, 1);
    zero_f<<<2, 256, 0, stream>>>(hist, 512);
    vq_gather<<<64, 256, 0, stream>>>(z2, (const float*)d_in[46], Idx, z3, vqacc, hist, 16384, 64); // zq = z3

    // ================= Decoder =================
    launch_conv(stream, z3, w_din, nullptr, nullptr, dA, nullptr, NB, 64, 64, 64, 64, 0, 128, 128, 0, 1, 1, 1, 0, 1, 1.f, 0);
    RB(dA, 64, 64, 128, 128, 128, 1, drw[0], drw[1], nullptr, nullptr, dB, dC, 128, 0);
    RB(dC, 64, 64, 128, 128, 128, 1, drw[2], drw[3], nullptr, nullptr, dB, dA, 128, 0);
    launch_conv(stream, dA, w_dt1, (const float*)d_in[53], nullptr, a2, nullptr,
                NB, 64, 64, 128, 128, 0, 64, 64, 0, 4, 4, 2, 2, 2, 1.f, 2);   // SiLU
    launch_conv(stream, a2, w_dt2, (const float*)d_in[55], nullptr, a1, nullptr,
                NB, 128, 128, 64, 64, 0, 32, 32, 0, 4, 4, 2, 2, 2, 1.f, 2);   // SiLU
    launch_conv(stream, a1, w_dt3, (const float*)d_in[57], nullptr, c32, nullptr,
                NB, 256, 256, 32, 32, 0, 16, 32, 16, 4, 4, 2, 2, 2, 1.f, 0);  // dec -> c32[:,16:32]

    // ================= Remixer =================
    launch_conv(stream, xh, w_rm1, nullptr, nullptr, r16, nullptr, NB, 512, 512, 3, 3, 0, 16, 16, 0, 1, 1, 1, 0, 1, 1.f, 1);
    launch_conv(stream, r16, w_rm2, nullptr, nullptr, c32, nullptr, NB, 512, 512, 16, 16, 0, 16, 32, 0, 1, 1, 1, 0, 1, 1.f, 1); // r -> c32[:,0:16]
    launch_conv(stream, c32, w_rm3, nullptr, nullptr, r32, nullptr, NB, 512, 512, 32, 32, 0, 32, 32, 0, 1, 1, 1, 0, 1, 1.f, 1);
    launch_conv(stream, r32, w_rm4, nullptr, nullptr, nullptr, (float*)d_out,
                NB, 512, 512, 32, 32, 0, 16, 16, 0, 1, 1, 1, 0, 1, 1.f, 1);   // recon, f32 NCHW

    vq_final<<<1, 512, 0, stream>>>(vqacc, hist, (float*)d_out, out_size, 16384, 64, 512);
}
